// TemporalItemGAT_17910013624757
// MI455X (gfx1250) — matrix operand order repaired and checked by reading, not compiled, not run on hardware
//
#include <hip/hip_runtime.h>
#include <hip/hip_bf16.h>

// ---------------------------------------------------------------------------
// TemporalItemGAT on gfx1250 (MI455X): 2-layer GAT.
//  - Dense projections: V_WMMA_F32_16X16X32_F16, pre-swizzled B fragments in
//    LDS (staged via global_load_async_to_lds_b128 + s_wait_asynccnt).
//  - Edge softmax/aggregation: L2-resident f32 atomics, 32 lanes per edge.
//  - Node buffers padded to multiples of 16 rows -> unguarded WMMA tile I/O.
// ---------------------------------------------------------------------------

typedef __attribute__((ext_vector_type(16))) _Float16 v16h;
typedef __attribute__((ext_vector_type(8)))  _Float16 v8h;
typedef __attribute__((ext_vector_type(2)))  _Float16 h2v;
typedef __attribute__((ext_vector_type(8)))  float    v8f;

#define NEG_SLOPE 0.2f

__device__ __forceinline__ float atomicMaxF(float* addr, float val) {
  unsigned int* ua = (unsigned int*)addr;
  unsigned int old = *ua;
  while (__uint_as_float(old) < val) {
    unsigned int assumed = old;
    old = atomicCAS(ua, assumed, __float_as_uint(val));
    if (old == assumed) break;
  }
  return __uint_as_float(old);
}

// Async-stage `bytes` from global into LDS (ws). 16B per lane per issue,
// full-EXEC iterations; completion via ASYNCcnt + s_wait_asynccnt.
// The flat-cast LDS pointer is passed into the asm as an (unused) operand so
// the __shared__ object's address escapes: combined with the "memory"
// clobber, the compiler must assume the asm writes `ws`, keeping the
// subsequent ds_load_b128 fragment reads live (round-2 bug: loads of a
// provably-never-stored LDS object were folded to undef).
__device__ __forceinline__ void stage_lds_async(_Float16* ws, const void* gsrc, int bytes) {
  const char* g = (const char*)gsrc;
  unsigned long long wsflat = (unsigned long long)(size_t)ws;   // escape LDS object
  for (int o = threadIdx.x * 16; o < bytes; o += blockDim.x * 16) {
    unsigned lds_off = (unsigned)o;                    // single shared var -> base 0
    unsigned long long ga = (unsigned long long)(g + o);
    asm volatile("global_load_async_to_lds_b128 %0, %1, off"
                 :: "v"(lds_off), "v"(ga), "v"(wsflat) : "memory");
  }
  asm volatile("s_wait_asynccnt 0x0" ::: "memory");
  __syncthreads();
}

// ------------------------------- init / prep -------------------------------

__global__ void k_fill_f32(float* __restrict__ p, long long n, float v) {
  long long i = (long long)blockIdx.x * blockDim.x + threadIdx.x;
  if (i < n) p[i] = v;
}

// x = emb[x_ids] with padding_idx=0 zeroed; rows [N, Npad) zero-filled.
__global__ void k_build_x(_Float16* __restrict__ xh, const int* __restrict__ x_ids,
                          const float* __restrict__ emb, int N, int Npad) {
  int i = blockIdx.x * blockDim.x + threadIdx.x;  // over Npad*32
  if (i >= Npad * 32) return;
  int row = i >> 5, col = i & 31;
  float v = 0.0f;
  if (row < N) {
    int id = x_ids[row];
    v = (id == 0) ? 0.0f : emb[(size_t)id * 32 + col];
  }
  xh[i] = (_Float16)v;
}

// Pre-swizzle W[K, Ncols] (f32) into WMMA B-fragment order (f16):
//   frag[((kt*NT+nt)*32 + lane)*16 + idx] = W[kt*32 + (lane>>4)*8 + (idx&7) + (idx>>3)*16][nt*16 + (lane&15)]
__global__ void k_swz_w(const float* __restrict__ W, _Float16* __restrict__ frag,
                        int KT, int NT, int Ncols) {
  int t = blockIdx.x * blockDim.x + threadIdx.x;
  int total = KT * NT * 512;
  if (t >= total) return;
  int idx  = t & 15;
  int lane = (t >> 4) & 31;
  int f    = t >> 9;
  int nt = f % NT, kt = f / NT;
  int k   = kt * 32 + (lane >> 4) * 8 + (idx & 7) + (idx >> 3) * 16;
  int col = nt * 16 + (lane & 15);
  frag[t] = (_Float16)W[(size_t)k * Ncols + col];
}

// ------------------------------- WMMA GEMMs --------------------------------
// A 16x32 f16 (per CDNA5 ISA): lane's 16 halves are two contiguous
// 8-half chunks of the row: K in [kb, kb+8) and [kb+16, kb+24), kb = hi*8.

__device__ __forceinline__ v16h load_a_frag(const _Float16* xr, int kb) {
  v8h lo = *(const v8h*)(xr + kb);
  v8h hi = *(const v8h*)(xr + kb + 16);
  return __builtin_shufflevector(lo, hi, 0, 1, 2, 3, 4, 5, 6, 7,
                                 8, 9, 10, 11, 12, 13, 14, 15);
}

// h1l[Npad,128] = xh[Npad,32] @ W1 (KT=1, NT=8)
__global__ void k_gemm1(const _Float16* __restrict__ xh, const _Float16* __restrict__ wfrag,
                        _Float16* __restrict__ h1l, int Npad) {
  __shared__ __align__(32) _Float16 ws[1 * 8 * 512];   // 8 KB fragment store
  stage_lds_async(ws, wfrag, 1 * 8 * 512 * 2);

  int wave = blockIdx.x * (blockDim.x >> 5) + (threadIdx.x >> 5);
  int lane = threadIdx.x & 31;
  int m0 = wave * 16;
  if (m0 >= Npad) return;

  int lrow = lane & 15;
  int hi = lane >> 4;
  const _Float16* xr = xh + (size_t)(m0 + lrow) * 32;
  v16h a = load_a_frag(xr, hi * 8);

  const v16h* bf = (const v16h*)ws;
#pragma unroll
  for (int nt = 0; nt < 8; ++nt) {
    v16h b = bf[nt * 32 + lane];                        // one 32B LDS read
    v8f c = {};
    c = __builtin_amdgcn_wmma_f32_16x16x32_f16(false, a, false, b, (short)0, c, false, false);
#pragma unroll
    for (int i = 0; i < 8; ++i)
      h1l[(size_t)(m0 + i + hi * 8) * 128 + nt * 16 + lrow] = (_Float16)c[i];
  }
}

// h2l[Npad,64] = h1f[Npad,128] @ W2 (KT=4, NT=4, chained accumulate)
__global__ void k_gemm2(const _Float16* __restrict__ h1f, const _Float16* __restrict__ wfrag,
                        _Float16* __restrict__ h2l, int Npad) {
  __shared__ __align__(32) _Float16 ws[4 * 4 * 512];   // 16 KB fragment store
  stage_lds_async(ws, wfrag, 4 * 4 * 512 * 2);

  int wave = blockIdx.x * (blockDim.x >> 5) + (threadIdx.x >> 5);
  int lane = threadIdx.x & 31;
  int m0 = wave * 16;
  if (m0 >= Npad) return;

  int lrow = lane & 15;
  int hi = lane >> 4;
  const _Float16* xr = h1f + (size_t)(m0 + lrow) * 128;

  v16h a[4];
#pragma unroll
  for (int ks = 0; ks < 4; ++ks) a[ks] = load_a_frag(xr, ks * 32 + hi * 8);

  const v16h* bf = (const v16h*)ws;
#pragma unroll
  for (int nt = 0; nt < 4; ++nt) {
    v8f c = {};
#pragma unroll
    for (int ks = 0; ks < 4; ++ks) {
      v16h b = bf[(ks * 4 + nt) * 32 + lane];
      c = __builtin_amdgcn_wmma_f32_16x16x32_f16(false, a[ks], false, b, (short)0, c, false, false);
    }
#pragma unroll
    for (int i = 0; i < 8; ++i)
      h2l[(size_t)(m0 + i + hi * 8) * 64 + nt * 16 + lrow] = (_Float16)c[i];
  }
}

// ---------------------------- attention scalars ----------------------------
__global__ void k_att(const _Float16* __restrict__ hl, const float* __restrict__ att_src,
                      const float* __restrict__ att_dst, float* __restrict__ asv,
                      float* __restrict__ adv, int N, int H, int C) {
  int t = blockIdx.x * blockDim.x + threadIdx.x;
  if (t >= N * H) return;
  int n = t / H, h = t % H;
  const _Float16* hp = hl + (size_t)n * H * C + (size_t)h * C;
  float s = 0.f, d = 0.f;
  for (int c0 = 0; c0 < C; c0 += 8) {
    v8h v = *(const v8h*)(hp + c0);
#pragma unroll
    for (int j = 0; j < 8; ++j) {
      float f = (float)v[j];
      s += f * att_src[h * C + c0 + j];
      d += f * att_dst[h * C + c0 + j];
    }
  }
  asv[t] = s; adv[t] = d;
}

// ------------------------------- edge passes -------------------------------

__device__ __forceinline__ void edge_ends(const int* __restrict__ eidx, int E, int e,
                                          int& s, int& d) {
  if (e < E) { s = eidx[e]; d = eidx[(size_t)E + e]; }
  else       { s = d = e - E; }           // appended self-loops
}

__global__ void k_edge_score(const int* __restrict__ eidx, int E, int Ep, int H,
                             const float* __restrict__ asv, const float* __restrict__ adv,
                             float* __restrict__ ebuf, float* __restrict__ mmax) {
  int t = blockIdx.x * blockDim.x + threadIdx.x;
  if (t >= Ep * H) return;
  int e = t / H, h = t % H;
  __builtin_prefetch(eidx + e + 4096, 0, 1);
  int s, d; edge_ends(eidx, E, e, s, d);
  float v = asv[s * H + h] + adv[d * H + h];
  v = v > 0.f ? v : NEG_SLOPE * v;
  ebuf[t] = v;
  atomicMaxF(&mmax[d * H + h], v);
}

__global__ void k_fix_m(float* __restrict__ m, int n) {
  int t = blockIdx.x * blockDim.x + threadIdx.x;
  if (t >= n) return;
  float v = m[t];
  if (!(v > -1e30f)) m[t] = 0.0f;
}

__global__ void k_edge_exp(const int* __restrict__ eidx, int E, int Ep, int H,
                           const float* __restrict__ mmax, float* __restrict__ ebuf,
                           float* __restrict__ ssum) {
  int t = blockIdx.x * blockDim.x + threadIdx.x;
  if (t >= Ep * H) return;
  int e = t / H, h = t % H;
  int s, d; edge_ends(eidx, E, e, s, d); (void)s;
  float ex = __expf(ebuf[t] - mmax[d * H + h]);
  ebuf[t] = ex;
  atomicAdd(&ssum[d * H + h], ex);
}

// agg[dst] += h[src] * alpha; 32 lanes/edge, each lane handles 2 packed f16
// channels per head (C == 64 exactly covered by 32 lanes x 2).
__global__ void k_edge_aggr(const int* __restrict__ eidx, int E, int Ep, int H, int C,
                            const _Float16* __restrict__ hl, const float* __restrict__ ebuf,
                            const float* __restrict__ ssum, float* __restrict__ agg) {
  long long gid = (long long)blockIdx.x * blockDim.x + threadIdx.x;
  int e = (int)(gid >> 5);
  int lane = (int)(gid & 31);
  if (e >= Ep) return;
  __builtin_prefetch(eidx + e + 1024, 0, 1);
  int s, d; edge_ends(eidx, E, e, s, d);
  int HC = H * C;
  const _Float16* hs = hl + (size_t)s * HC;
  float* ad = agg + (size_t)d * HC;
  for (int h = 0; h < H; ++h) {
    float alpha = ebuf[(size_t)e * H + h] / (ssum[d * H + h] + 1e-16f);
    int c0 = h * C + 2 * lane;
    h2v p = *(const h2v*)(hs + c0);
    atomicAdd(&ad[c0],     (float)p[0] * alpha);
    atomicAdd(&ad[c0 + 1], (float)p[1] * alpha);
  }
}

// ------------------------------- finalizers --------------------------------

// h1f = f16(relu(agg1 + bias1)); pad rows [N, Npad) zeroed for GEMM2 A-reads.
__global__ void k_final1(const float* __restrict__ agg, const float* __restrict__ bias,
                         _Float16* __restrict__ h1f, int N, int Npad) {
  int t = blockIdx.x * blockDim.x + threadIdx.x;
  if (t >= Npad * 128) return;
  int row = t >> 7;
  float v = 0.0f;
  if (row < N) {
    v = agg[t] + bias[t & 127];
    v = v > 0.f ? v : 0.f;
  }
  h1f[t] = (_Float16)v;
}

__global__ void k_final2(float* __restrict__ out, const float* __restrict__ bias, int N) {
  int t = blockIdx.x * blockDim.x + threadIdx.x;
  if (t >= N * 64) return;
  out[t] = out[t] + bias[t & 63];
}

// ------------------------------- launch glue -------------------------------

static inline int gridFor(long long n, int b) { return (int)((n + b - 1) / b); }

extern "C" void kernel_launch(void* const* d_in, const int* in_sizes, int n_in,
                              void* d_out, int out_size, void* d_ws, size_t ws_size,
                              hipStream_t stream) {
  const int*   x_ids    = (const int*)d_in[0];
  const int*   eidx     = (const int*)d_in[1];
  const float* item_emb = (const float*)d_in[2];
  const float* W1       = (const float*)d_in[3];
  const float* att_src1 = (const float*)d_in[4];
  const float* att_dst1 = (const float*)d_in[5];
  const float* bias1    = (const float*)d_in[6];
  const float* W2       = (const float*)d_in[7];
  const float* att_src2 = (const float*)d_in[8];
  const float* att_dst2 = (const float*)d_in[9];
  const float* bias2    = (const float*)d_in[10];
  float* out = (float*)d_out;

  const int N    = in_sizes[0];
  const int E    = in_sizes[1] / 2;
  const int Ep   = E + N;                    // with self-loops
  const int Npad = ((N + 15) / 16) * 16;     // WMMA row padding

  // -------- workspace carving (256B aligned) --------
  size_t off = 0;
  char* base = (char*)d_ws;
  auto carve = [&](size_t bytes) -> void* {
    off = (off + 255) & ~(size_t)255;
    void* p = base + off;
    off += bytes;
    return p;
  };
  _Float16* xh    = (_Float16*)carve((size_t)Npad * 32 * 2);
  _Float16* w1f   = (_Float16*)carve((size_t)1 * 8 * 512 * 2);   // W1 fragments
  _Float16* w2f   = (_Float16*)carve((size_t)4 * 4 * 512 * 2);   // W2 fragments
  _Float16* h1l   = (_Float16*)carve((size_t)Npad * 128 * 2);
  _Float16* h1f   = (_Float16*)carve((size_t)Npad * 128 * 2);
  _Float16* h2l   = (_Float16*)carve((size_t)Npad * 64 * 2);
  float* asv1 = (float*)carve((size_t)N * 2 * 4);
  float* adv1 = (float*)carve((size_t)N * 2 * 4);
  float* m1   = (float*)carve((size_t)N * 2 * 4);
  float* s1   = (float*)carve((size_t)N * 2 * 4);
  float* asv2 = (float*)carve((size_t)N * 4);
  float* adv2 = (float*)carve((size_t)N * 4);
  float* m2   = (float*)carve((size_t)N * 4);
  float* s2   = (float*)carve((size_t)N * 4);
  float* agg1 = (float*)carve((size_t)N * 128 * 4);
  float* eb1  = (float*)carve((size_t)Ep * 2 * 4);
  float* eb2  = (float*)carve((size_t)Ep * 4);
  (void)ws_size; (void)n_in; (void)out_size;

  const int B = 256;
  const float NEG_INF = -__builtin_inff();

  // -------- prep --------
  k_build_x<<<gridFor((long long)Npad * 32, B), B, 0, stream>>>(xh, x_ids, item_emb, N, Npad);
  k_swz_w<<<gridFor(1 * 8 * 512, B), B, 0, stream>>>(W1, w1f, 1, 8, 128);
  k_swz_w<<<gridFor(4 * 4 * 512, B), B, 0, stream>>>(W2, w2f, 4, 4, 64);
  k_fill_f32<<<gridFor((long long)N * 2, B), B, 0, stream>>>(m1, (long long)N * 2, NEG_INF);
  k_fill_f32<<<gridFor((long long)N * 2, B), B, 0, stream>>>(s1, (long long)N * 2, 0.f);
  k_fill_f32<<<gridFor((long long)N, B), B, 0, stream>>>(m2, (long long)N, NEG_INF);
  k_fill_f32<<<gridFor((long long)N, B), B, 0, stream>>>(s2, (long long)N, 0.f);
  k_fill_f32<<<gridFor((long long)N * 128, B), B, 0, stream>>>(agg1, (long long)N * 128, 0.f);
  k_fill_f32<<<gridFor((long long)N * 64, B), B, 0, stream>>>(out, (long long)N * 64, 0.f);

  const int rowBlocks = Npad / 16;
  const int wavesPerBlk = B / 32;

  // -------- layer 1 --------
  k_gemm1<<<gridFor(rowBlocks, wavesPerBlk), B, 0, stream>>>(xh, w1f, h1l, Npad);
  k_att<<<gridFor((long long)N * 2, B), B, 0, stream>>>(h1l, att_src1, att_dst1, asv1, adv1, N, 2, 64);
  k_edge_score<<<gridFor((long long)Ep * 2, B), B, 0, stream>>>(eidx, E, Ep, 2, asv1, adv1, eb1, m1);
  k_fix_m<<<gridFor((long long)N * 2, B), B, 0, stream>>>(m1, N * 2);
  k_edge_exp<<<gridFor((long long)Ep * 2, B), B, 0, stream>>>(eidx, E, Ep, 2, m1, eb1, s1);
  k_edge_aggr<<<gridFor((long long)Ep * 32, B), B, 0, stream>>>(eidx, E, Ep, 2, 64, h1l, eb1, s1, agg1);
  k_final1<<<gridFor((long long)Npad * 128, B), B, 0, stream>>>(agg1, bias1, h1f, N, Npad);

  // -------- layer 2 --------
  k_gemm2<<<gridFor(rowBlocks, wavesPerBlk), B, 0, stream>>>(h1f, w2f, h2l, Npad);
  k_att<<<gridFor((long long)N, B), B, 0, stream>>>(h2l, att_src2, att_dst2, asv2, adv2, N, 1, 64);
  k_edge_score<<<gridFor((long long)Ep, B), B, 0, stream>>>(eidx, E, Ep, 1, asv2, adv2, eb2, m2);
  k_fix_m<<<gridFor((long long)N, B), B, 0, stream>>>(m2, N);
  k_edge_exp<<<gridFor((long long)Ep, B), B, 0, stream>>>(eidx, E, Ep, 1, m2, eb2, s2);
  k_edge_aggr<<<gridFor((long long)Ep * 32, B), B, 0, stream>>>(eidx, E, Ep, 1, 64, h2l, eb2, s2, out);
  k_final2<<<gridFor((long long)N * 64, B), B, 0, stream>>>(out, bias2, N);
}
